// TransferModel_28441273434614
// MI455X (gfx1250) — matrix operand order repaired and checked by reading, not compiled, not run on hardware
//
#include <hip/hip_runtime.h>
#include <hip/hip_bf16.h>
#include <stdint.h>
#include <math.h>

// ---------------------------------------------------------------- types
typedef __attribute__((ext_vector_type(16))) _Float16 v16h;
typedef __attribute__((ext_vector_type(8)))  _Float16 v8h;
typedef __attribute__((ext_vector_type(8)))  float    v8f;
typedef __attribute__((ext_vector_type(2)))  float    v2f;
typedef __attribute__((ext_vector_type(4)))  unsigned int u32x4;
typedef __attribute__((ext_vector_type(8)))  int      i32x8;
typedef __attribute__((ext_vector_type(4)))  int      i32x4;

#define ALPHA_C   0.6f
#define NCLUST    3
#define LAMBD     0.1f
#define GAMMA_C   0.01f
#define KM_ITERS  10
#define NS_ITERS  12
#define FC        512     // channels at f4
#define FHW       1024    // 32x32 spatial at f4
#define CONV_KC   32

// ------------------------------------------------- TDM helper (2D f16 tile)
// Builds a D# per CDNA5 ISA and issues
// tensor_load_to_lds.  Arity differs between ROCm7.2 (5 args) and
// clang-23/therock (6 args); guard on __clang_major__.
__device__ __forceinline__ void tdm_load_tile_f16(void* ldsPtr, const void* gptr,
                                                  unsigned rows, unsigned cols,
                                                  unsigned rowStrideElems) {
#if __has_builtin(__builtin_amdgcn_tensor_load_to_lds)
  unsigned lds_addr = (unsigned)(uintptr_t)ldsPtr;           // LDS aperture: addr[31:0]
  unsigned long long ga = (unsigned long long)(uintptr_t)gptr;
  u32x4 g0;
  g0[0] = 1u;                                                // count=1, user descriptor
  g0[1] = lds_addr;                                          // lds_addr [63:32]
  g0[2] = (unsigned)ga;                                      // global_addr lo
  g0[3] = ((unsigned)(ga >> 32) & 0x01FFFFFFu) | (2u << 30); // global_addr hi, type=2
  i32x8 g1;
  g1[0] = 1 << 16;                                           // wg_mask=0, data_size=1 (2B)
  g1[1] = (int)((cols & 0xFFFFu) << 16);                     // tensor_dim0 [15:0]
  g1[2] = (int)(((cols >> 16) & 0xFFFFu) | ((rows & 0xFFFFu) << 16)); // dim0 hi | dim1 lo
  g1[3] = (int)(((rows >> 16) & 0xFFFFu) | ((cols & 0xFFFFu) << 16)); // dim1 hi | tile_dim0
  g1[4] = (int)(rows & 0xFFFFu);                             // tile_dim1 | tile_dim2=0
  g1[5] = (int)rowStrideElems;                               // tensor_dim0_stride lo32
  g1[6] = 0;
  g1[7] = 0;
  i32x4 zz = {0, 0, 0, 0};
#if defined(__clang_major__) && (__clang_major__ >= 23)
  i32x8 z8 = {0, 0, 0, 0, 0, 0, 0, 0};
  __builtin_amdgcn_tensor_load_to_lds(g0, g1, zz, zz, z8, 0);
#else
  __builtin_amdgcn_tensor_load_to_lds(g0, g1, zz, zz, 0);
#endif
#else
  // Fallback: plain copy (single-lane caller)
  const _Float16* s = (const _Float16*)gptr;
  _Float16* d = (_Float16*)ldsPtr;
  for (unsigned r = 0; r < rows; ++r)
    for (unsigned c = 0; c < cols; ++c)
      d[r * cols + c] = s[r * rowStrideElems + c];
#endif
}

__device__ __forceinline__ void tdm_wait() {
#if __has_builtin(__builtin_amdgcn_s_wait_tensorcnt)
  __builtin_amdgcn_s_wait_tensorcnt(0);
#endif
}

// ---------------------------------------------------------- weight packing
// wp[(kt*KP + n)*32 + kc] = w[n][c][dy][dx] as f16, kglobal = kt*32+kc = c*9+t
__global__ void pack_w_kernel(const float* __restrict__ w, _Float16* __restrict__ wp,
                              int K, int Cc, int KP, int KdimP) {
  long i = (long)blockIdx.x * blockDim.x + threadIdx.x;
  long total = (long)KdimP * KP;
  if (i >= total) return;
  int kc = (int)(i & 31);
  long t = i >> 5;
  int n = (int)(t % KP);
  int kt = (int)(t / KP);
  int kg = kt * 32 + kc;
  float v = 0.0f;
  int Kdim = Cc * 9;
  if (n < K && kg < Kdim) {
    int c = kg / 9, tt = kg - c * 9;
    v = w[(((size_t)n * Cc + c) * 3 + tt / 3) * 3 + (tt % 3)];
  }
  wp[i] = (_Float16)v;
}

// --------------------------------------------- conv3x3 implicit-GEMM (f16 WMMA)
// M = H*W pixels, N = output channels, Kdim = C*9.  128x64 tile per workgroup,
// 8 waves each own a 16(M)x64(N) strip -> 4 WMMA accumulators per wave; the A
// fragment is loaded once per K-slice and reused across 4 B fragments.
// B tile staged by TDM; A tile gathered (3x3 halo + f32->f16) into LDS.
__global__ __launch_bounds__(256)
void conv3x3_wmma(const float* __restrict__ in, const _Float16* __restrict__ wp,
                  const float* __restrict__ bias, float* __restrict__ out,
                  int C, int K, int H, int W, int relu, int KP) {
  __shared__ __align__(32) _Float16 lA[128 * CONV_KC];   // 8 KB
  __shared__ __align__(32) _Float16 lB[64 * CONV_KC];    // 4 KB
  const int HW = H * W;
  const int Kdim = C * 9;
  const int KdimP = (Kdim + CONV_KC - 1) / CONV_KC * CONV_KC;
  const int m0 = blockIdx.x * 128;
  const int n0 = blockIdx.y * 64;
  const int b  = blockIdx.z;
  const int lane = threadIdx.x & 31;
  const int wave = threadIdx.x >> 5;       // 0..7 : 16-row M chunk
  v8f acc[4] = {{}, {}, {}, {}};
  const int mA  = threadIdx.x >> 1;        // 0..127 : pixel row of A tile
  const int kA0 = (threadIdx.x & 1) * 16;  // 0 or 16
  const int pix = m0 + mA;
  const int py = pix / W, px = pix % W;

  for (int k0 = 0; k0 < KdimP; k0 += CONV_KC) {
    const _Float16* wt = wp + ((size_t)(k0 / CONV_KC) * KP + n0) * CONV_KC;
    if (threadIdx.x == 0)
      tdm_load_tile_f16(lB, wt, 64, CONV_KC, CONV_KC);
    if (threadIdx.x == 32 && (k0 + CONV_KC) < KdimP)
      __builtin_prefetch(wp + ((size_t)(k0 / CONV_KC + 1) * KP + n0) * CONV_KC, 0, 3);

    #pragma unroll
    for (int e = 0; e < 16; ++e) {
      int kg = k0 + kA0 + e;
      float v = 0.0f;
      if (kg < Kdim) {
        int c = kg / 9, t = kg - c * 9;
        int yy = py + t / 3 - 1, xx = px + (t % 3) - 1;
        if (yy >= 0 && yy < H && xx >= 0 && xx < W)
          v = in[(((size_t)b * C + c) * H + yy) * W + xx];
      }
      lA[mA * CONV_KC + kA0 + e] = (_Float16)v;
    }
    tdm_wait();
    __syncthreads();

    // A fragment: lane L -> row M = L%16; K = 16*(L/16)..+15
    const int mrow = wave * 16 + (lane & 15);
    const int ka   = 16 * (lane >> 4);
    union { v16h v; v8h h[2]; } ua;
    ua.h[0] = *(const v8h*)&lA[mrow * CONV_KC + ka];
    ua.h[1] = *(const v8h*)&lA[mrow * CONV_KC + 8 + ka];
    // B fragments: lane L -> col N = L%16 (+16t); K = 16*(L/16)..+15
    const int koff = 16 * (lane >> 4);
    #pragma unroll
    for (int t = 0; t < 4; ++t) {
      int nn = t * 16 + (lane & 15);
      v16h bf = *(const v16h*)&lB[nn * CONV_KC + koff];
      acc[t] = __builtin_amdgcn_wmma_f32_16x16x32_f16(false, ua.v, false, bf,
                                                      (short)0, acc[t], false, false);
    }
    __syncthreads();
  }

  #pragma unroll
  for (int t = 0; t < 4; ++t) {
    #pragma unroll
    for (int r = 0; r < 8; ++r) {
      int mpix = m0 + wave * 16 + r + 8 * (lane >> 4);
      int n = n0 + t * 16 + (lane & 15);
      if (n < K) {
        float v = acc[t][r] + bias[n];
        if (relu) v = fmaxf(v, 0.0f);
        out[((size_t)b * K + n) * HW + mpix] = v;
      }
    }
  }
}

// -------------------------------------------------------------- pool / upsample
__global__ void pool2_kernel(const float* __restrict__ in, float* __restrict__ out,
                             int H, int W, long total) {
  long i = (long)blockIdx.x * blockDim.x + threadIdx.x;
  if (i >= total) return;
  int Wo = W >> 1, Ho = H >> 1;
  int xo = (int)(i % Wo); long t = i / Wo;
  int yo = (int)(t % Ho); long bc = t / Ho;
  const float* p = in + (bc * H + 2 * yo) * W + 2 * xo;
  out[i] = 0.25f * (p[0] + p[1] + p[W] + p[W + 1]);
}

__global__ void up2_kernel(const float* __restrict__ in, float* __restrict__ out,
                           int Ho, int Wo, long total) {
  long i = (long)blockIdx.x * blockDim.x + threadIdx.x;
  if (i >= total) return;
  int x = (int)(i % Wo); long t = i / Wo;
  int y = (int)(t % Ho); long bc = t / Ho;
  out[i] = in[(bc * (Ho >> 1) + (y >> 1)) * (Wo >> 1) + (x >> 1)];
}

// -------------------------------------------------------------------- k-means
__global__ __launch_bounds__(1024)
void kmeans_kernel(const float* __restrict__ cf, const float* __restrict__ sf,
                   int* __restrict__ asg_c, int* __restrict__ asg_s,
                   float* __restrict__ mu_c, float* __restrict__ mu_s,
                   float* __restrict__ cnt_c, float* __restrict__ cnt_s) {
  __shared__ float cent[NCLUST * FC];
  __shared__ float cnts[NCLUST];
  __shared__ unsigned char asg[FHW];
  int z = blockIdx.x; int src = z & 1; int b = z >> 1;
  const float* f = (src ? sf : cf) + (size_t)b * FC * FHW;
  int tid = threadIdx.x;
  for (int i = tid; i < NCLUST * FC; i += blockDim.x) {
    int kk = i / FC, c = i - kk * FC;
    cent[i] = f[(size_t)c * FHW + kk];          // init = first NCLUST points
  }
  __syncthreads();
  for (int it = 0; it <= KM_ITERS; ++it) {
    if (tid < FHW) {
      float best = 3.4e38f; int bk = 0;
      for (int kk = 0; kk < NCLUST; ++kk) {
        float d = 0.0f;
        for (int c = 0; c < FC; ++c) {
          float t = f[(size_t)c * FHW + tid] - cent[kk * FC + c];
          d += t * t;
        }
        if (d < best) { best = d; bk = kk; }
      }
      asg[tid] = (unsigned char)bk;
    }
    __syncthreads();
    for (int i = tid; i < NCLUST * FC; i += blockDim.x) {
      int kk = i / FC, c = i - kk * FC;
      float s = 0.0f, n = 0.0f;
      for (int j = 0; j < FHW; ++j)
        if (asg[j] == kk) { s += f[(size_t)c * FHW + j]; n += 1.0f; }
      cent[i] = s / (n + 1e-6f);
      if (c == 0) cnts[kk] = n;
    }
    __syncthreads();
  }
  if (tid < FHW) (src ? asg_s : asg_c)[b * FHW + tid] = (int)asg[tid];
  float* muo = (src ? mu_s : mu_c) + (size_t)b * NCLUST * FC;
  for (int i = tid; i < NCLUST * FC; i += blockDim.x) muo[i] = cent[i];
  if (tid < NCLUST) (src ? cnt_s : cnt_c)[b * NCLUST + tid] = cnts[tid];
}

// ------------------------------------- weighted covariance via f32 WMMA 16x16x4
// cov[z] = (f-mu)*w @ (f-mu)^T / nc + LAMBD*I ; z = ((b*3+k)*2 + src), 24 total
// 128x64 tile, wave = 16x64 strip, 4 accumulators.
__global__ __launch_bounds__(256)
void cov_wmma(const float* __restrict__ cf, const float* __restrict__ sf,
              const int* __restrict__ asg_c, const int* __restrict__ asg_s,
              const float* __restrict__ mu_c, const float* __restrict__ mu_s,
              const float* __restrict__ cnt_c, const float* __restrict__ cnt_s,
              float* __restrict__ cov) {
  int z = blockIdx.z;
  int src = z & 1; int bk = z >> 1; int b = bk / NCLUST; int k = bk - b * NCLUST;
  const float* f  = (src ? sf : cf) + (size_t)b * FC * FHW;
  const int*  asg = (src ? asg_s : asg_c) + b * FHW;
  const float* mu = (src ? mu_s : mu_c) + ((size_t)b * NCLUST + k) * FC;
  float nc = (src ? cnt_s : cnt_c)[b * NCLUST + k] + 1e-6f;
  int m0 = blockIdx.y * 128, n0 = blockIdx.x * 64;
  int lane = threadIdx.x & 31, wave = threadIdx.x >> 5, g = lane >> 4;
  int mrow = m0 + wave * 16 + (lane & 15);
  float mum = mu[mrow];
  int ncl[4]; float mun[4];
  #pragma unroll
  for (int t = 0; t < 4; ++t) { ncl[t] = n0 + t * 16 + (lane & 15); mun[t] = mu[ncl[t]]; }
  v8f acc[4] = {{}, {}, {}, {}};
  for (int j0 = 0; j0 < FHW; j0 += 4) {
    int ja = j0 + 2 * g;
    float w0 = (asg[ja] == k) ? 1.0f : 0.0f;
    float w1 = (asg[ja + 1] == k) ? 1.0f : 0.0f;
    v2f a;  a[0] = (f[(size_t)mrow * FHW + ja] - mum) * w0;
            a[1] = (f[(size_t)mrow * FHW + ja + 1] - mum) * w1;
    #pragma unroll
    for (int t = 0; t < 4; ++t) {
      v2f bb; bb[0] = f[(size_t)ncl[t] * FHW + ja] - mun[t];
              bb[1] = f[(size_t)ncl[t] * FHW + ja + 1] - mun[t];
      acc[t] = __builtin_amdgcn_wmma_f32_16x16x4_f32(false, a, false, bb,
                                                     (short)0, acc[t], false, false);
    }
  }
  float inv = 1.0f / nc;
  float* Cz = cov + (size_t)z * FC * FC;
  #pragma unroll
  for (int t = 0; t < 4; ++t) {
    #pragma unroll
    for (int r = 0; r < 8; ++r) {
      int mm = m0 + wave * 16 + r + 8 * g;
      int nn = n0 + t * 16 + (lane & 15);
      Cz[(size_t)mm * FC + nn] = acc[t][r] * inv + ((mm == nn) ? LAMBD : 0.0f);
    }
  }
}

// ---------------------------------------- generic batched f32 WMMA GEMM (NN)
// C[z] = aeff * A[z/zdivA] @ B[z/zdivB] + beta * C[z]; row-major,
// M mult of 128, N mult of 64, K mult of 4.  Wave = 16x64 strip, 4 accums.
// sclMode=1: aeff = alpha * (scl[2z+1]/scl[2z])^(1/4)  (Newton-Schulz scaling)
__global__ __launch_bounds__(256)
void gemm_f32_wmma(const float* __restrict__ A, const float* __restrict__ B,
                   float* __restrict__ Cm, int M, int N, int Kd,
                   long sA, long sB, long sC, int zdivA, int zdivB,
                   float alpha, float beta, const float* __restrict__ scl, int sclMode) {
  int z = blockIdx.z;
  const float* Az = A + (size_t)(z / zdivA) * sA;
  const float* Bz = B + (size_t)(z / zdivB) * sB;
  float* Cz = Cm + (size_t)z * sC;
  int m0 = blockIdx.y * 128, n0 = blockIdx.x * 64;
  int lane = threadIdx.x & 31, wave = threadIdx.x >> 5, g = lane >> 4;
  int mrow = m0 + wave * 16 + (lane & 15);
  int nl = n0 + (lane & 15);
  v8f acc[4] = {{}, {}, {}, {}};
  for (int k0 = 0; k0 < Kd; k0 += 4) {
    int ka = k0 + 2 * g;
    v2f a;  a[0] = Az[(size_t)mrow * Kd + ka];
            a[1] = Az[(size_t)mrow * Kd + ka + 1];
    #pragma unroll
    for (int t = 0; t < 4; ++t) {
      v2f bb; bb[0] = Bz[(size_t)ka * N + nl + t * 16];
              bb[1] = Bz[(size_t)(ka + 1) * N + nl + t * 16];
      acc[t] = __builtin_amdgcn_wmma_f32_16x16x4_f32(false, a, false, bb,
                                                     (short)0, acc[t], false, false);
    }
  }
  float aeff = alpha;
  if (sclMode == 1)
    aeff = alpha * sqrtf(sqrtf(scl[2 * z + 1] / fmaxf(scl[2 * z], 1e-30f)));
  #pragma unroll
  for (int t = 0; t < 4; ++t) {
    #pragma unroll
    for (int r = 0; r < 8; ++r) {
      int mm = m0 + wave * 16 + r + 8 * g;
      size_t i0 = (size_t)mm * N + n0 + t * 16 + (lane & 15);
      float c0 = (beta != 0.0f) ? Cz[i0] : 0.0f;
      Cz[i0] = aeff * acc[t][r] + beta * c0;
    }
  }
}

// ------------------------------------------------------- small elementwise kit
__global__ void zero_kernel(float* p, long n) {
  long i = (long)blockIdx.x * blockDim.x + threadIdx.x;
  if (i < n) p[i] = 0.0f;
}

__global__ void frob_kernel(const float* __restrict__ A, float* __restrict__ norms, long perz) {
  __shared__ float red[256];
  int z = blockIdx.z;
  long i = (long)blockIdx.x * 256 + threadIdx.x;
  float v = 0.0f;
  if (i < perz) { float a = A[(size_t)z * perz + i]; v = a * a; }
  red[threadIdx.x] = v; __syncthreads();
  for (int o = 128; o > 0; o >>= 1) {
    if (threadIdx.x < o) red[threadIdx.x] += red[threadIdx.x + o];
    __syncthreads();
  }
  if (threadIdx.x == 0) atomicAdd(&norms[z], red[0]);
}

__global__ void scale_invnorm_kernel(const float* __restrict__ A, float* __restrict__ Y,
                                     const float* __restrict__ norms, long perz) {
  int z = blockIdx.z;
  long i = (long)blockIdx.x * 256 + threadIdx.x;
  if (i >= perz) return;
  float nf = sqrtf(fmaxf(norms[z], 1e-30f));   // Frobenius norm
  Y[(size_t)z * perz + i] = A[(size_t)z * perz + i] / nf;
}

__global__ void ident_kernel(float* __restrict__ Z, int N, long perz) {
  int z = blockIdx.z;
  long i = (long)blockIdx.x * 256 + threadIdx.x;
  if (i >= perz) return;
  int r = (int)(i / N), c = (int)(i - (long)r * N);
  Z[(size_t)z * perz + i] = (r == c) ? 1.0f : 0.0f;
}

__global__ void t2_kernel(float* __restrict__ P, int N, long perz) {  // P <- 1.5I - 0.5P
  int z = blockIdx.z;
  long i = (long)blockIdx.x * 256 + threadIdx.x;
  if (i >= perz) return;
  int r = (int)(i / N), c = (int)(i - (long)r * N);
  size_t idx = (size_t)z * perz + i;
  P[idx] = ((r == c) ? 1.5f : 0.0f) - 0.5f * P[idx];
}

// v[z][c] = sum_d T[z][c][d] * mu_c[z][d]  (z = b*3+k)
__global__ void tmu_kernel(const float* __restrict__ T, const float* __restrict__ mu_c,
                           float* __restrict__ v) {
  int z = blockIdx.y;
  int c = blockIdx.x * blockDim.x + threadIdx.x;
  if (c >= FC) return;
  const float* Tz = T + ((size_t)z * FC + c) * FC;
  const float* mu = mu_c + (size_t)z * FC;
  float s = 0.0f;
  for (int d = 0; d < FC; ++d) s += Tz[d] * mu[d];
  v[(size_t)z * FC + c] = s;
}

// tf[b][c][j] = ALPHA*(U[b*3+k][c][j] - v[b*3+k][c] + mu_s[b][k][c]) + (1-ALPHA)*cf
__global__ void apply_kernel(const float* __restrict__ U, const float* __restrict__ v,
                             const float* __restrict__ mu_s, const int* __restrict__ asg_c,
                             const float* __restrict__ cf, float* __restrict__ tf, long total) {
  long i = (long)blockIdx.x * blockDim.x + threadIdx.x;
  if (i >= total) return;
  int j = (int)(i % FHW); long t = i / FHW;
  int c = (int)(t % FC); int b = (int)(t / FC);
  int k = asg_c[b * FHW + j];
  int z = b * NCLUST + k;
  float val = U[((size_t)z * FC + c) * FHW + j] - v[(size_t)z * FC + c]
            + mu_s[(size_t)z * FC + c];
  tf[i] = ALPHA_C * val + (1.0f - ALPHA_C) * cf[i];
}

// per-(b,c) mean / std over spatial
__global__ void chan_stats_kernel(const float* __restrict__ f, float* __restrict__ st,
                                  int HW, int BC) {
  int i = blockIdx.x * blockDim.x + threadIdx.x;
  if (i >= BC) return;
  const float* p = f + (size_t)i * HW;
  float s = 0.0f, s2 = 0.0f;
  for (int j = 0; j < HW; ++j) { float v = p[j]; s += v; s2 += v * v; }
  float m = s / HW;
  st[i] = m;
  st[BC + i] = sqrtf(fmaxf(s2 / HW - m * m, 0.0f));
}

__global__ void stats_mse_kernel(const float* __restrict__ se, const float* __restrict__ ss,
                                 float* __restrict__ outp, int BC) {
  __shared__ float red[256];
  int tid = threadIdx.x;
  float s = 0.0f;
  for (int i = tid; i < BC; i += 256) {
    float dm = se[i] - ss[i];
    float ds = se[BC + i] - ss[BC + i];
    s += dm * dm + ds * ds;
  }
  red[tid] = s; __syncthreads();
  for (int o = 128; o > 0; o >>= 1) {
    if (tid < o) red[tid] += red[tid + o];
    __syncthreads();
  }
  if (tid == 0) atomicAdd(outp, GAMMA_C * red[0] / BC);
}

__global__ void mse_kernel(const float* __restrict__ a, const float* __restrict__ b,
                           float* __restrict__ outp, long n) {
  __shared__ float red[256];
  float s = 0.0f;
  for (long i = (long)blockIdx.x * 256 + threadIdx.x; i < n; i += (long)gridDim.x * 256) {
    float d = a[i] - b[i];
    s += d * d;
  }
  red[threadIdx.x] = s; __syncthreads();
  for (int o = 128; o > 0; o >>= 1) {
    if (threadIdx.x < o) red[threadIdx.x] += red[threadIdx.x + o];
    __syncthreads();
  }
  if (threadIdx.x == 0) atomicAdd(outp, red[0] / (float)n);
}

// =========================================================== host orchestration
extern "C" void kernel_launch(void* const* d_in, const int* in_sizes, int n_in,
                              void* d_out, int out_size, void* d_ws, size_t ws_size,
                              hipStream_t stream) {
  (void)in_sizes; (void)n_in; (void)out_size; (void)ws_size;
  const int B = 4;
  const float* cimg = (const float*)d_in[0];
  const float* simg = (const float*)d_in[1];
  const float* Wt[8]  = {(const float*)d_in[2],  (const float*)d_in[4],
                         (const float*)d_in[6],  (const float*)d_in[8],
                         (const float*)d_in[10], (const float*)d_in[12],
                         (const float*)d_in[14], (const float*)d_in[16]};
  const float* Bi[8]  = {(const float*)d_in[3],  (const float*)d_in[5],
                         (const float*)d_in[7],  (const float*)d_in[9],
                         (const float*)d_in[11], (const float*)d_in[13],
                         (const float*)d_in[15], (const float*)d_in[17]};
  static const int Cs[8] = {3, 64, 128, 256, 512, 256, 128, 64};
  static const int Ks[8] = {64, 128, 256, 512, 256, 128, 64, 3};

  char* base = (char*)d_ws;
  size_t off = 0;
  auto alloc = [&](size_t bytes) -> void* {
    void* p = base + off;
    off += (bytes + 255) & ~(size_t)255;
    return p;
  };

  _Float16* wp[8]; int KPa[8], KdPa[8];
  for (int l = 0; l < 8; ++l) {
    KPa[l]  = (Ks[l] + 63) / 64 * 64;
    KdPa[l] = (Cs[l] * 9 + 31) / 32 * 32;
    wp[l]   = (_Float16*)alloc((size_t)KPa[l] * KdPa[l] * sizeof(_Float16));
  }
  const size_t ACT = (size_t)B * 64 * 256 * 256;   // largest activation
  float* act0 = (float*)alloc(ACT * 4);
  float* act1 = (float*)alloc(ACT * 4);
  const size_t F4 = (size_t)B * FC * FHW;
  float* cf  = (float*)alloc(F4 * 4);
  float* sfb = (float*)alloc(F4 * 4);
  float* tf  = (float*)alloc(F4 * 4);
  float* ef  = (float*)alloc(F4 * 4);
  float* dec = (float*)alloc((size_t)B * 3 * 256 * 256 * 4);
  int*   asg_c = (int*)alloc(B * FHW * 4);
  int*   asg_s = (int*)alloc(B * FHW * 4);
  float* mu_c  = (float*)alloc((size_t)B * NCLUST * FC * 4);
  float* mu_s  = (float*)alloc((size_t)B * NCLUST * FC * 4);
  float* cnt_c = (float*)alloc(B * NCLUST * 4);
  float* cnt_s = (float*)alloc(B * NCLUST * 4);
  const long N2 = (long)FC * FC;
  float* cov = (float*)alloc((size_t)24 * N2 * 4);   // also reused as NS "P"
  float* Ya  = (float*)alloc((size_t)24 * N2 * 4);
  float* Yb  = (float*)alloc((size_t)24 * N2 * 4);
  float* Za  = (float*)alloc((size_t)24 * N2 * 4);
  float* Zb  = (float*)alloc((size_t)24 * N2 * 4);
  float* norms = (float*)alloc(24 * 4);
  float* T   = (float*)alloc((size_t)12 * N2 * 4);
  float* U   = (float*)alloc((size_t)12 * FC * FHW * 4);
  float* vkc = (float*)alloc((size_t)12 * FC * 4);
  int Cl[4] = {64, 128, 256, 512};
  float* statsS[4]; float* statsE[4];
  for (int l = 0; l < 4; ++l) {
    statsS[l] = (float*)alloc((size_t)2 * B * Cl[l] * 4);
    statsE[l] = (float*)alloc((size_t)2 * B * Cl[l] * 4);
  }

  // ---- pack weights to WMMA B layout
  for (int l = 0; l < 8; ++l) {
    long total = (long)KdPa[l] * KPa[l];
    pack_w_kernel<<<(unsigned)((total + 255) / 256), 256, 0, stream>>>(
        Wt[l], wp[l], Ks[l], Cs[l], KPa[l], KdPa[l]);
  }

  auto conv = [&](const float* in, int l, float* out, int H, int Wd, int relu) {
    dim3 g((H * Wd) / 128, KPa[l] / 64, B);
    conv3x3_wmma<<<g, 256, 0, stream>>>(in, wp[l], Bi[l], out, Cs[l], Ks[l], H, Wd, relu, KPa[l]);
  };
  auto pool = [&](const float* in, float* out, int Cn, int H, int Wd) {
    long total = (long)B * Cn * (H / 2) * (Wd / 2);
    pool2_kernel<<<(unsigned)((total + 255) / 256), 256, 0, stream>>>(in, out, H, Wd, total);
  };
  auto up = [&](const float* in, float* out, int Cn, int Ho, int Wo) {
    long total = (long)B * Cn * Ho * Wo;
    up2_kernel<<<(unsigned)((total + 255) / 256), 256, 0, stream>>>(in, out, Ho, Wo, total);
  };
  auto stats = [&](const float* f, float* st, int Cn, int HW) {
    int BC = B * Cn;
    chan_stats_kernel<<<(BC + 255) / 256, 256, 0, stream>>>(f, st, HW, BC);
  };
  auto encode = [&](const float* img, float* outF, float* const* st) {
    conv(img, 0, act0, 256, 256, 1);
    if (st) stats(act0, st[0], 64, 65536);
    pool(act0, act1, 64, 256, 256);
    conv(act1, 1, act0, 128, 128, 1);
    if (st) stats(act0, st[1], 128, 16384);
    pool(act0, act1, 128, 128, 128);
    conv(act1, 2, act0, 64, 64, 1);
    if (st) stats(act0, st[2], 256, 4096);
    pool(act0, act1, 256, 64, 64);
    conv(act1, 3, outF, 32, 32, 1);
    if (st) stats(outF, st[3], 512, 1024);
  };

  // ---- encode content and style (style keeps per-layer stats)
  encode(cimg, cf, nullptr);
  encode(simg, sfb, statsS);

  // ---- clustered WCT: kmeans -> covariances -> Newton-Schulz +/-1/2 -> T
  kmeans_kernel<<<B * 2, 1024, 0, stream>>>(cf, sfb, asg_c, asg_s, mu_c, mu_s, cnt_c, cnt_s);
  {
    dim3 g(FC / 64, FC / 128, 24);
    cov_wmma<<<g, 256, 0, stream>>>(cf, sfb, asg_c, asg_s, mu_c, mu_s, cnt_c, cnt_s, cov);
  }
  zero_kernel<<<1, 32, 0, stream>>>(norms, 24);
  {
    dim3 g((unsigned)((N2 + 255) / 256), 1, 24);
    frob_kernel<<<g, 256, 0, stream>>>(cov, norms, N2);
    scale_invnorm_kernel<<<g, 256, 0, stream>>>(cov, Ya, norms, N2);
    ident_kernel<<<g, 256, 0, stream>>>(Za, FC, N2);
  }
  float* Yc = Ya; float* Yn = Yb; float* Zc = Za; float* Zn = Zb; float* P = cov;
  for (int it = 0; it < NS_ITERS; ++it) {
    dim3 g(FC / 64, FC / 128, 24);
    dim3 ge((unsigned)((N2 + 255) / 256), 1, 24);
    gemm_f32_wmma<<<g, 256, 0, stream>>>(Zc, Yc, P, FC, FC, FC, N2, N2, N2, 1, 1,
                                         1.0f, 0.0f, nullptr, 0);
    t2_kernel<<<ge, 256, 0, stream>>>(P, FC, N2);                 // P <- 1.5I - 0.5 ZY
    gemm_f32_wmma<<<g, 256, 0, stream>>>(Yc, P, Yn, FC, FC, FC, N2, N2, N2, 1, 1,
                                         1.0f, 0.0f, nullptr, 0); // Y <- Y P
    gemm_f32_wmma<<<g, 256, 0, stream>>>(P, Zc, Zn, FC, FC, FC, N2, N2, N2, 1, 1,
                                         1.0f, 0.0f, nullptr, 0); // Z <- P Z
    float* t0 = Yc; Yc = Yn; Yn = t0;
    float* t1 = Zc; Zc = Zn; Zn = t1;
  }
  // T[z] = sqrt(cov_s) @ invsqrt(cov_c) = scale * Y_style @ Z_content, z = b*3+k
  {
    dim3 g(FC / 64, FC / 128, 12);
    gemm_f32_wmma<<<g, 256, 0, stream>>>(Yc + N2, Zc, T, FC, FC, FC,
                                         2 * N2, 2 * N2, N2, 1, 1,
                                         1.0f, 0.0f, norms, 1);
  }
  // U[z] = T[z] @ f_content[b]
  {
    dim3 g(FHW / 64, FC / 128, 12);
    gemm_f32_wmma<<<g, 256, 0, stream>>>(T, cf, U, FC, FHW, FC,
                                         N2, (long)FC * FHW, (long)FC * FHW, 1, 3,
                                         1.0f, 0.0f, nullptr, 0);
  }
  tmu_kernel<<<dim3(FC / 256, 12), 256, 0, stream>>>(T, mu_c, vkc);
  {
    long total = (long)B * FC * FHW;
    apply_kernel<<<(unsigned)((total + 255) / 256), 256, 0, stream>>>(
        U, vkc, mu_s, asg_c, cf, tf, total);
  }

  // ---- decode
  conv(tf, 4, act0, 32, 32, 1);        // d1: 512->256 @32
  up(act0, act1, 256, 64, 64);
  conv(act1, 5, act0, 64, 64, 1);      // d2: 256->128 @64
  up(act0, act1, 128, 128, 128);
  conv(act1, 6, act0, 128, 128, 1);    // d3: 128->64 @128
  up(act0, act1, 64, 256, 256);
  conv(act1, 7, dec, 256, 256, 0);     // d4: 64->3 @256, no ReLU

  // ---- re-encode decoded image (with stats), losses
  encode(dec, ef, statsE);
  zero_kernel<<<1, 32, 0, stream>>>((float*)d_out, 1);
  mse_kernel<<<256, 256, 0, stream>>>(ef, cf, (float*)d_out, (long)F4);
  for (int l = 0; l < 4; ++l)
    stats_mse_kernel<<<1, 256, 0, stream>>>(statsE[l], statsS[l], (float*)d_out, B * Cl[l]);
}
